// SkipLSTM_20547123544854
// MI455X (gfx1250) — matrix operand order repaired and checked by reading, not compiled, not run on hardware
//
#include <hip/hip_runtime.h>
#include <hip/hip_bf16.h>
#include <stdint.h>

// SkipLSTM for MI455X (gfx1250): per-step GEMMs in v_wmma_f32_16x16x32_bf16
// (bf16 operands, f32 accumulate), gates/state in f32. Weights converted once
// to bf16 column-major so B fragments are single contiguous 32B loads.
// GEMM tiling: 32M x 64N per wave (8 WMMA tiles), 1.5 b128-loads per WMMA.

#define BSZ   128
#define SEQL  200
#define NCODE 20
#define ICH   256
#define HCH   512
#define ZN    2048   // 4*HCH

typedef __attribute__((ext_vector_type(16))) __bf16 v16bf;
typedef __attribute__((ext_vector_type(8)))  __bf16 v8bf;
typedef __attribute__((ext_vector_type(8)))  float  v8f;

__device__ __forceinline__ unsigned short f2bf(float x) {
  unsigned int u = __float_as_uint(x);
  u += 0x7FFFu + ((u >> 16) & 1u);     // round-to-nearest-even
  return (unsigned short)(u >> 16);
}
__device__ __forceinline__ float sigm(float x) { return 1.0f / (1.0f + expf(-x)); }

// ---- W (K x 2048, f32, row-major) -> Wt (2048 x K, bf16, row-major) ----
__global__ void wconv_kernel(const float* __restrict__ W, unsigned short* __restrict__ Wt, int K) {
  int total = K * ZN;
  for (int i = blockIdx.x * blockDim.x + threadIdx.x; i < total; i += gridDim.x * blockDim.x) {
    int n = i / K, k = i - n * K;
    Wt[i] = f2bf(W[(size_t)k * ZN + n]);
  }
}

// ---- embedding bag: xe_bf16[(t*BS+b)*IC + ic] = sum_codes emb[x[b,t,code], ic] ----
__global__ void embed_kernel(const int* __restrict__ x, const float* __restrict__ emb,
                             unsigned short* __restrict__ xeb) {
  int tb = blockIdx.x;               // t*BSZ + b
  int t = tb / BSZ, b = tb - t * BSZ;
  int ic = threadIdx.x;              // 256 threads
  const int* xp = x + ((size_t)b * SEQL + t) * NCODE;
  float acc = 0.f;
#pragma unroll
  for (int cde = 0; cde < NCODE; ++cde)
    acc += emb[(size_t)xp[cde] * ICH + ic];
  xeb[(size_t)tb * ICH + ic] = f2bf(acc);
}

__global__ void init_kernel(float* h0, float* c0, unsigned short* h0b,
                            float* h1, float* c1, unsigned short* h1b,
                            float* u0, float* u1, float* du, int* skip) {
  int i = blockIdx.x * blockDim.x + threadIdx.x;
  if (i < BSZ * HCH) {
    h0[i] = 0.f; c0[i] = 0.f; h1[i] = 0.f; c1[i] = 0.f;
    h0b[i] = 0;  h1b[i] = 0;
  }
  if (i < BSZ) { u0[i] = 1.f; u1[i] = 1.f; du[i] = 0.f; skip[i] = 0; }
}

// ---- per-step GEMM: z(128x2048,f32) = [A1|A2](128xK,bf16) @ Wt^T + bias ----
// Grid: 16 blocks (128 cols each) x 256 threads = 8 waves arranged 4(M) x 2(N).
// Each wave: 32 rows x 64 cols = 2x4 accumulator tiles; K-loop in chunks of 32,
// split into the A1 region and the A2 region (both K1,K2 are multiples of 32).
__global__ __launch_bounds__(256) void gemm_kernel(
    const unsigned short* __restrict__ A1, int K1,
    const unsigned short* __restrict__ A2, int K2,
    const unsigned short* __restrict__ Wt,
    const float* __restrict__ bias, float* __restrict__ z) {
  const int K    = K1 + K2;
  const int wave = threadIdx.x >> 5;
  const int lane = threadIdx.x & 31;
  const int l16  = lane & 15;
  const int hi   = lane >> 4;
  const int wm   = wave & 3;          // M quarter: rows [wm*32, wm*32+32)
  const int wn   = wave >> 2;         // N half within block
  const int ncol0 = blockIdx.x * 128 + wn * 64;

  v8f acc[2][4];
#pragma unroll
  for (int m = 0; m < 2; ++m)
#pragma unroll
    for (int s = 0; s < 4; ++s)
      acc[m][s] = (v8f){0.f,0.f,0.f,0.f,0.f,0.f,0.f,0.f};

  // A row base pointers for the wave's two M-tiles.
  const int r0 = wm * 32 + l16;
  const __bf16* wb = reinterpret_cast<const __bf16*>(Wt);

#pragma unroll
  for (int part = 0; part < 2; ++part) {
    const __bf16* Ap = reinterpret_cast<const __bf16*>(part == 0 ? A1 : A2);
    const int Kp     = (part == 0) ? K1 : K2;
    const int koff   = (part == 0) ? 0 : K1;   // offset into Wt's k dimension
    const __bf16* arow0 = Ap + (size_t)r0 * Kp;
    const __bf16* arow1 = Ap + (size_t)(r0 + 16) * Kp;
    const int nchunks = Kp >> 5;
    for (int kc = 0; kc < nchunks; ++kc) {
      const int kb = kc << 5;
      // A fragment, this code's contiguous k order: lane = row M=l16 (+hi selects k-halves);
      // lanes 0-15 k in [0,16), lanes 16-31 k in [16,32) (matches B).
      v16bf afrag[2];
#pragma unroll
      for (int m = 0; m < 2; ++m) {
        const __bf16* ap = (m == 0 ? arow0 : arow1) + kb;
        afrag[m] = *(const v16bf*)(ap + hi * 16);
      }
      // B fragment, same contiguous k order as A: lane = col N=l16; lanes 0-15 k=kb..kb+15,
      // lanes 16-31 k=kb+16..kb+31 -> one contiguous 32B load from Wt[n].
#pragma unroll
      for (int s = 0; s < 4; ++s) {
        const int n = ncol0 + s * 16 + l16;
        const v16bf bfrag = *(const v16bf*)(wb + (size_t)n * K + koff + kb + hi * 16);
#pragma unroll
        for (int m = 0; m < 2; ++m)
          acc[m][s] = __builtin_amdgcn_wmma_f32_16x16x32_bf16(
              false, afrag[m], false, bfrag, (short)0, acc[m][s], false, false);
      }
    }
  }

  // C/D layout: lanes 0-15: N=lane, M=r; lanes 16-31: N=lane-16, M=8+r.
#pragma unroll
  for (int s = 0; s < 4; ++s) {
    const int col = ncol0 + s * 16 + l16;
    const float bc = bias[col];
#pragma unroll
    for (int m = 0; m < 2; ++m) {
      const int rowb = wm * 32 + m * 16 + hi * 8;
#pragma unroll
      for (int r = 0; r < 8; ++r)
        z[(size_t)(rowb + r) * ZN + col] = acc[m][s][r] + bc;
    }
  }
}

// ---- per-step cell: one block per batch row, 256 threads x 2 hidden units ----
__global__ __launch_bounds__(256) void cell_kernel(
    const float* __restrict__ z, float* __restrict__ h, float* __restrict__ c,
    unsigned short* __restrict__ hb, unsigned short* __restrict__ hs_out,
    float* __restrict__ u, float* __restrict__ du, int* __restrict__ skip,
    const float* __restrict__ pw, const float* __restrict__ pb) {
  __shared__ float red[256];
  const int b = blockIdx.x;
  const int tid = threadIdx.x;
  const float u_b = u[b], du_b = du[b];
  const int skip_b = skip[b];
  const float bu = rintf(u_b);         // round half-to-even, matches jnp.round
  const float one_m = 1.f - bu;

  float ncv[2], nhv[2], cv[2], hv[2];
  float partial = 0.f;
#pragma unroll
  for (int q = 0; q < 2; ++q) {
    const int j = tid + q * 256;
    const float* zr = z + (size_t)b * ZN;
    const float f = zr[j], ig = zr[HCH + j], o = zr[2 * HCH + j], g = zr[3 * HCH + j];
    const float cc = c[(size_t)b * HCH + j];
    const float hh = h[(size_t)b * HCH + j];
    const float nc = cc * sigm(f + 1.0f) + tanhf(g) * sigm(ig);
    const float nh = sigm(o) * tanhf(nc);
    ncv[q] = nc; nhv[q] = nh; cv[q] = cc; hv[q] = hh;
    partial += nc * bu * pw[j];        // (nc*bu) . pw
  }
  red[tid] = partial;
  __syncthreads();
  for (int st = 128; st > 0; st >>= 1) {
    if (tid < st) red[tid] += red[tid + st];
    __syncthreads();
  }
  const float du_n  = sigm(red[0] + pb[0]);
  const float nu_n  = du_n * bu;
  const float nu_s  = fminf(fmaxf(u_b + du_b, 0.f), 1.f) * one_m;
  const float new_u = skip_b ? nu_s : nu_n;
  const float new_du = skip_b ? du_b : du_n;
  const float n_after = ceilf(0.5f / fmaxf(new_u, 1e-12f)) - 1.0f;
  if (tid == 0) { u[b] = new_u; du[b] = new_du; skip[b] = (n_after > 0.f) ? 1 : 0; }
#pragma unroll
  for (int q = 0; q < 2; ++q) {
    const int j = tid + q * 256;
    const float new_h = skip_b ? hv[q] * one_m : nhv[q] * bu;
    const float new_c = skip_b ? cv[q] * one_m : ncv[q] * bu;
    h[(size_t)b * HCH + j] = new_h;
    c[(size_t)b * HCH + j] = new_c;
    const unsigned short hb16 = f2bf(new_h);
    hb[(size_t)b * HCH + j] = hb16;
    if (hs_out) hs_out[(size_t)b * HCH + j] = hb16;
  }
}

// ---- out = h1_last(128x512,f32) @ Wout(512x2) + bout ----
__global__ void out_kernel(const float* __restrict__ h1, const float* __restrict__ Wout,
                           const float* __restrict__ bout, float* __restrict__ out) {
  const int tid = threadIdx.x;   // 256 = 128 rows x 2 cols
  const int b = tid >> 1, jo = tid & 1;
  float acc = bout[jo];
  for (int k = 0; k < HCH; ++k)
    acc += h1[(size_t)b * HCH + k] * Wout[k * 2 + jo];
  out[b * 2 + jo] = acc;
}

extern "C" void kernel_launch(void* const* d_in, const int* in_sizes, int n_in,
                              void* d_out, int out_size, void* d_ws, size_t ws_size,
                              hipStream_t stream) {
  const int*   x    = (const int*)  d_in[0];
  const float* emb  = (const float*)d_in[5];
  const float* W0   = (const float*)d_in[6];
  const float* b0   = (const float*)d_in[7];
  const float* pw0  = (const float*)d_in[8];
  const float* pb0  = (const float*)d_in[9];
  const float* W1   = (const float*)d_in[10];
  const float* b1   = (const float*)d_in[11];
  const float* pw1  = (const float*)d_in[12];
  const float* pb1  = (const float*)d_in[13];
  const float* Wout = (const float*)d_in[14];
  const float* bout = (const float*)d_in[15];
  float* out = (float*)d_out;

  char* ws = (char*)d_ws;
  size_t off = 0;
  auto take = [&](size_t bytes) -> char* {
    char* p = ws + off;
    off = (off + bytes + 255) & ~(size_t)255;
    return p;
  };
  unsigned short* wt0  = (unsigned short*)take((size_t)ZN * 768 * 2);   // 3.0 MB
  unsigned short* wt1  = (unsigned short*)take((size_t)ZN * 1024 * 2);  // 4.0 MB
  unsigned short* xeb  = (unsigned short*)take((size_t)SEQL * BSZ * ICH * 2);  // 13.1 MB
  unsigned short* hs0b = (unsigned short*)take((size_t)SEQL * BSZ * HCH * 2);  // 26.2 MB
  float* zbuf = (float*)take((size_t)BSZ * ZN * 4);                     // 1.0 MB
  float* h0 = (float*)take((size_t)BSZ * HCH * 4);
  float* c0 = (float*)take((size_t)BSZ * HCH * 4);
  unsigned short* h0b = (unsigned short*)take((size_t)BSZ * HCH * 2);
  float* h1 = (float*)take((size_t)BSZ * HCH * 4);
  float* c1 = (float*)take((size_t)BSZ * HCH * 4);
  unsigned short* h1b = (unsigned short*)take((size_t)BSZ * HCH * 2);
  float* u0  = (float*)take(BSZ * 4);
  float* u1  = (float*)take(BSZ * 4);
  float* duv = (float*)take(BSZ * 4);
  int*   skp = (int*)  take(BSZ * 4);

  wconv_kernel<<<1024, 256, 0, stream>>>(W0, wt0, 768);
  wconv_kernel<<<1024, 256, 0, stream>>>(W1, wt1, 1024);
  embed_kernel<<<SEQL * BSZ, ICH, 0, stream>>>(x, emb, xeb);
  init_kernel<<<(BSZ * HCH + 255) / 256, 256, 0, stream>>>(h0, c0, h0b, h1, c1, h1b,
                                                           u0, u1, duv, skp);
  // Layer 0 scan (skip/du carry forward into layer 1 afterwards)
  for (int t = 0; t < SEQL; ++t) {
    gemm_kernel<<<16, 256, 0, stream>>>(xeb + (size_t)t * BSZ * ICH, ICH, h0b, HCH,
                                        wt0, b0, zbuf);
    cell_kernel<<<BSZ, 256, 0, stream>>>(zbuf, h0, c0, h0b,
                                         hs0b + (size_t)t * BSZ * HCH,
                                         u0, duv, skp, pw0, pb0);
  }
  // Layer 1 scan
  for (int t = 0; t < SEQL; ++t) {
    gemm_kernel<<<16, 256, 0, stream>>>(hs0b + (size_t)t * BSZ * HCH, HCH, h1b, HCH,
                                        wt1, b1, zbuf);
    cell_kernel<<<BSZ, 256, 0, stream>>>(zbuf, h1, c1, h1b, nullptr,
                                         u1, duv, skp, pw1, pb1);
  }
  out_kernel<<<1, 256, 0, stream>>>(h1, Wout, bout, out);
}
